// GDNTokenShiftBlock_36910948942183
// MI455X (gfx1250) — compile-verified, matrix operand order checked by reading, not run on hardware
//
#include <hip/hip_runtime.h>

// ---------------------------------------------------------------------------
// Gated DeltaNet block for MI455X (gfx1250, wave32, WMMA bf16 16x16x32,
// async global->LDS staging for GEMM weight tiles)
// ---------------------------------------------------------------------------

typedef __bf16 bf16;
typedef __attribute__((ext_vector_type(16))) __bf16 bf16x16;
typedef __attribute__((ext_vector_type(8)))  float  f32x8;

constexpr int NB   = 4;      // batch
constexpr int TT   = 4096;   // sequence length
constexpr int DD   = 1024;   // model dim
constexpr int NH   = 8;      // heads
constexpr int KK   = 128;    // head dim
constexpr int CC   = 64;     // chunk size
constexpr int NCH  = TT / CC;        // 64 chunks
constexpr int BTOK = NB * TT;        // 16384 tokens

__device__ __forceinline__ bf16 to_bf16(float x) { return (bf16)x; }
__device__ __forceinline__ bf16 to_bf16(bf16 x)  { return x; }

__device__ __forceinline__ f32x8 f32x8_zero() {
  f32x8 z;
#pragma unroll
  for (int e = 0; e < 8; ++e) z[e] = 0.f;
  return z;
}

__device__ __forceinline__ f32x8 wmma_bf16(bf16x16 a, bf16x16 b, f32x8 c) {
  // D = A(16x32) * B(32x16) + C   -> v_wmma_f32_16x16x32_bf16
  return __builtin_amdgcn_wmma_f32_16x16x32_bf16(
      /*neg_a=*/false, a, /*neg_b=*/false, b,
      /*c_mod=*/(short)0, c, /*reuse_a=*/false, /*reuse_b=*/false);
}

// Async DMA of 64 contiguous bytes global->LDS per lane (4x b128); the
// instruction immediate offset is added to BOTH the LDS and global address
// (CDNA5 ISA 10.x async pseudocode), so one base pair covers all four.
__device__ __forceinline__ void async_copy_b128x4(void* lds_dst,
                                                  const void* gsrc) {
  unsigned lds_addr = (unsigned)(uintptr_t)lds_dst;  // low 32 bits = LDS offset
  asm volatile(
      "global_load_async_to_lds_b128 %0, %1, off\n\t"
      "global_load_async_to_lds_b128 %0, %1, off offset:16\n\t"
      "global_load_async_to_lds_b128 %0, %1, off offset:32\n\t"
      "global_load_async_to_lds_b128 %0, %1, off offset:48"
      :: "v"(lds_addr), "v"(gsrc) : "memory");
}
__device__ __forceinline__ void wait_async0() {
  asm volatile("s_wait_asynccnt 0x0" ::: "memory");
}

// ---- fragment loaders (wave32 VGPR layouts per CDNA5 ISA 7.12.2) -----------
// A 16x32 bf16: lane L holds row m=L&15; half=L>>4 selects K sub-blocks.
template <typename T>
__device__ __forceinline__ bf16x16 frag_a(const T* base, int ld, int lane) {
  int m = lane & 15, h2 = lane >> 4;
  bf16x16 f;
#pragma unroll
  for (int i = 0; i < 8; ++i) {
    int k = (i >> 2) * 16 + h2 * 8 + (i & 3) * 2;
    f[2 * i]     = to_bf16(base[m * ld + k]);
    f[2 * i + 1] = to_bf16(base[m * ld + k + 1]);
  }
  return f;
}

// A fragment with per-row scale (used for rk * exp(decay))
__device__ __forceinline__ bf16x16 frag_a_scaled(const bf16* base, int ld,
                                                 int lane, float s) {
  int m = lane & 15, h2 = lane >> 4;
  bf16x16 f;
#pragma unroll
  for (int i = 0; i < 8; ++i) {
    int k = (i >> 2) * 16 + h2 * 8 + (i & 3) * 2;
    f[2 * i]     = to_bf16((float)base[m * ld + k] * s);
    f[2 * i + 1] = to_bf16((float)base[m * ld + k + 1] * s);
  }
  return f;
}

// A[m][c] = wk[c][m] * dw[c]  (transposed read + per-k scale) for state update
__device__ __forceinline__ bf16x16 frag_a_wkT(const bf16* wkc, int m0, int c0,
                                              int lane, const float* dw) {
  int m = m0 + (lane & 15), h2 = lane >> 4;
  bf16x16 f;
#pragma unroll
  for (int i = 0; i < 8; ++i) {
    int k = (i >> 2) * 16 + h2 * 8 + (i & 3) * 2;
    int c = c0 + k;
    f[2 * i]     = to_bf16((float)wkc[(size_t)c * KK + m] * dw[c]);
    f[2 * i + 1] = to_bf16((float)wkc[(size_t)(c + 1) * KK + m] * dw[c + 1]);
  }
  return f;
}

// B 32x16: lane L holds column n=L&15; same K striping as A.
template <typename T>
__device__ __forceinline__ bf16x16 frag_b(const T* base, int ld, int lane) {
  int n = lane & 15, h2 = lane >> 4;
  bf16x16 f;
#pragma unroll
  for (int i = 0; i < 8; ++i) {
    int k = (i >> 2) * 16 + h2 * 8 + (i & 3) * 2;
    f[2 * i]     = to_bf16(base[(size_t)k * ld + n]);
    f[2 * i + 1] = to_bf16(base[(size_t)(k + 1) * ld + n]);
  }
  return f;
}

// B[k][n] = src[n][k] (transposed source, e.g. wk^T)
template <typename T>
__device__ __forceinline__ bf16x16 frag_bt(const T* base, int ld, int lane) {
  int n = lane & 15, h2 = lane >> 4;
  bf16x16 f;
#pragma unroll
  for (int i = 0; i < 8; ++i) {
    int k = (i >> 2) * 16 + h2 * 8 + (i & 3) * 2;
    f[2 * i]     = to_bf16(base[(size_t)n * ld + k]);
    f[2 * i + 1] = to_bf16(base[(size_t)n * ld + k + 1]);
  }
  return f;
}

// C/D 16x16 f32: lane 0-15 holds N=lane; VGPR r holds M = 8*half + r
__device__ __forceinline__ void store_c_f32(float* base, int ld, int lane,
                                            f32x8 c) {
  int n = lane & 15, h2 = lane >> 4;
#pragma unroll
  for (int r = 0; r < 8; ++r) base[(size_t)(h2 * 8 + r) * ld + n] = c[r];
}
__device__ __forceinline__ void store_c_bf(bf16* base, int ld, int lane,
                                           f32x8 c) {
  int n = lane & 15, h2 = lane >> 4;
#pragma unroll
  for (int r = 0; r < 8; ++r)
    base[(size_t)(h2 * 8 + r) * ld + n] = to_bf16(c[r]);
}
template <typename T>
__device__ __forceinline__ f32x8 load_c(const T* base, int ld, int lane) {
  int n = lane & 15, h2 = lane >> 4;
  f32x8 c;
#pragma unroll
  for (int r = 0; r < 8; ++r) c[r] = (float)base[(size_t)(h2 * 8 + r) * ld + n];
  return c;
}
__device__ __forceinline__ f32x8 load_c_scaled(const float* base, int ld,
                                               int lane, float s) {
  int n = lane & 15, h2 = lane >> 4;
  f32x8 c;
#pragma unroll
  for (int r = 0; r < 8; ++r)
    c[r] = base[(size_t)(h2 * 8 + r) * ld + n] * s;
  return c;
}

// ---------------------------------------------------------------------------
// K1: pack Wv/Wg into a fused transposed bf16 B-matrix [K=1024, N=2048],
//     and Wo into [1024,1024].
// ---------------------------------------------------------------------------
__global__ __launch_bounds__(256) void pack_weights_kernel(
    const float* __restrict__ Wv, const float* __restrict__ Wg,
    const float* __restrict__ Wo, bf16* __restrict__ WvgT,
    bf16* __restrict__ WoT) {
  int idx = blockIdx.x * 256 + threadIdx.x;
  if (idx >= DD * DD) return;
  int n = idx / DD, k = idx % DD;  // W[n][k] -> B[k][n]
  WvgT[(size_t)k * 2048 + n]        = to_bf16(Wv[idx]);
  WvgT[(size_t)k * 2048 + 1024 + n] = to_bf16(Wg[idx]);
  WoT[(size_t)k * 1024 + n]         = to_bf16(Wo[idx]);
}

// ---------------------------------------------------------------------------
// K2/K8: GEMM C(f32) = A(f32|bf16) * B(bf16).
// Block tile 128x256, 8 waves in a 2x4 grid, each wave owns a 64x64 patch
// (16 WMMA accumulators -> 16 WMMAs per 8 fragment loads per K-step).
// B tiles are staged via async global->LDS DMA (ASYNCcnt).
// ---------------------------------------------------------------------------
template <typename TA>
__global__ __launch_bounds__(256) void gemm_bf16_kernel(
    const TA* __restrict__ A, const bf16* __restrict__ Bm,
    float* __restrict__ Cm, int M, int N, int Kd) {
  __shared__ bf16 As[128 * 32];   // 8 KB
  __shared__ bf16 Bs[32 * 256];   // 16 KB
  int bm = blockIdx.x, bn = blockIdx.y;
  int tid = threadIdx.x, lane = tid & 31, wave = tid >> 5;
  int wm = wave >> 2, wn = wave & 3;  // 2x4 wave grid, 64x64 per wave

  f32x8 acc[4][4];
#pragma unroll
  for (int i = 0; i < 4; ++i)
#pragma unroll
    for (int j = 0; j < 4; ++j) acc[i][j] = f32x8_zero();

  int ar  = tid >> 1, ac0 = (tid & 1) * 16;   // A loader: 16 elems/thread
  int bkr = tid >> 3, bc0 = (tid & 7) * 32;   // B loader: 32 elems/thread
  int nk = Kd / 32;
  for (int ks = 0; ks < nk; ++ks) {
    int k0 = ks * 32;
    const TA* ap = A + (size_t)(bm * 128 + ar) * Kd + k0 + ac0;
    const bf16* bp = Bm + (size_t)(k0 + bkr) * N + bn * 256 + bc0;
    // B tile: async DMA straight into LDS, tracked by ASYNCcnt
    async_copy_b128x4(&Bs[bkr * 256 + bc0], bp);
    // A tile: VALU staging (fp32 -> bf16 conversion on the fly)
#pragma unroll
    for (int e = 0; e < 16; ++e) As[ar * 32 + ac0 + e] = to_bf16(ap[e]);
    if (ks + 1 < nk) {  // hint next A tile into cache (global_prefetch_b8)
      __builtin_prefetch(ap + 32, 0, 0);
    }
    wait_async0();
    __syncthreads();
    bf16x16 af[4], fb[4];
#pragma unroll
    for (int i = 0; i < 4; ++i)
      af[i] = frag_a(&As[(wm * 64 + i * 16) * 32], 32, lane);
#pragma unroll
    for (int j = 0; j < 4; ++j)
      fb[j] = frag_b(&Bs[wn * 64 + j * 16], 256, lane);
#pragma unroll
    for (int i = 0; i < 4; ++i)
#pragma unroll
      for (int j = 0; j < 4; ++j) acc[i][j] = wmma_bf16(af[i], fb[j], acc[i][j]);
    __syncthreads();
  }
#pragma unroll
  for (int i = 0; i < 4; ++i)
#pragma unroll
    for (int j = 0; j < 4; ++j) {
      float* cp = Cm + (size_t)(bm * 128 + wm * 64 + i * 16) * N +
                  bn * 256 + wn * 64 + j * 16;
      store_c_f32(cp, N, lane, acc[i][j]);
    }
}

// ---------------------------------------------------------------------------
// K3: beta = sigmoid(x.Wb), decay = -exp(A_log)*softplus(x.Wa + dt_bias),
//     plus per-chunk cumulative decay. One block per (batch, chunk).
// ---------------------------------------------------------------------------
__global__ __launch_bounds__(256) void small_proj_kernel(
    const float* __restrict__ x, const float* __restrict__ Wb,
    const float* __restrict__ Wa, const float* __restrict__ dt_bias,
    const float* __restrict__ A_log, float* __restrict__ beta,
    float* __restrict__ deccum) {
  int b = blockIdx.x / NCH, ch = blockIdx.x % NCH;
  __shared__ float dec_s[NH][CC];
  int tid = threadIdx.x;
  for (int w = tid; w < CC * NH; w += 256) {
    int c = w >> 3, hh = w & 7;
    int t = ch * CC + c;
    const float* xr = x + ((size_t)b * TT + t) * DD;
    const float* wb = Wb + hh * DD;
    const float* wa = Wa + hh * DD;
    float db = 0.f, da = 0.f;
    for (int d = 0; d < DD; ++d) {
      float xv = xr[d];
      db += xv * wb[d];
      da += xv * wa[d];
    }
    beta[((size_t)b * TT + t) * NH + hh] = 1.f / (1.f + __expf(-db));
    float z  = da + dt_bias[hh];
    float sp = (z > 20.f) ? z : log1pf(__expf(z));
    dec_s[hh][c] = -__expf(A_log[hh]) * sp;
  }
  __syncthreads();
  if (tid < NH) {
    float run = 0.f;
    for (int c = 0; c < CC; ++c) {
      run += dec_s[tid][c];
      deccum[(((size_t)b * NH + tid) * NCH + ch) * CC + c] = run;
    }
  }
}

// ---------------------------------------------------------------------------
// K4: per token build chunked bf16 tensors:
//   rk = l2norm(x)/sqrt(K); wk = l2norm(x[t-1]);
//   v  = silu(conv(x.Wv)) * beta; wkb = wk*beta; u = wkb*exp(deccum)
// One block per token.
// ---------------------------------------------------------------------------
__global__ __launch_bounds__(256) void qkv_kernel(
    const float* __restrict__ x, const float* __restrict__ xvg,
    const float* __restrict__ conv_w, const float* __restrict__ conv_b,
    const float* __restrict__ beta, const float* __restrict__ deccum,
    bf16* __restrict__ rk, bf16* __restrict__ wk, bf16* __restrict__ v,
    bf16* __restrict__ wkb, bf16* __restrict__ u) {
  int tok = blockIdx.x;
  int b = tok / TT, t = tok % TT;
  __shared__ float ss[2 * NH];
  int tid = threadIdx.x;
  if (tid < 2 * NH) ss[tid] = 0.f;
  __syncthreads();
  int d0 = tid * 4;
  float xc[4], xp[4], pc = 0.f, pp = 0.f;
#pragma unroll
  for (int e = 0; e < 4; ++e) {
    int d = d0 + e;
    xc[e] = x[(size_t)tok * DD + d];
    xp[e] = (t > 0) ? x[((size_t)tok - 1) * DD + d] : 0.f;
    pc += xc[e] * xc[e];
    pp += xp[e] * xp[e];
  }
  int hh = d0 >> 7;
  atomicAdd(&ss[hh], pc);
  atomicAdd(&ss[NH + hh], pp);
  __syncthreads();
  float rn_c = rsqrtf(ss[hh] + 1e-6f) * 0.08838834764831845f;  // /sqrt(128)
  float rn_p = rsqrtf(ss[NH + hh] + 1e-6f);
  float bet = beta[(size_t)tok * NH + hh];
  int ch = t / CC, c = t % CC;
  float de = __expf(deccum[(((size_t)b * NH + hh) * NCH + ch) * CC + c]);
  size_t base = ((((size_t)b * NH + hh) * NCH + ch) * CC + c) * (size_t)KK;
#pragma unroll
  for (int e = 0; e < 4; ++e) {
    int d = d0 + e, k = d & 127;
    // causal depthwise conv over x.Wv (stored in xvg cols [0,1024))
    float acc = conv_b[d];
#pragma unroll
    for (int j = 0; j < 4; ++j) {
      int tt2 = t - 3 + j;
      float xvp = (tt2 >= 0) ? xvg[((size_t)b * TT + tt2) * 2048 + d] : 0.f;
      acc += xvp * conv_w[d * 4 + j];
    }
    float sil = acc / (1.f + __expf(-acc));
    float wkv = xp[e] * rn_p;
    rk[base + k]  = to_bf16(xc[e] * rn_c);
    wk[base + k]  = to_bf16(wkv);
    v[base + k]   = to_bf16(sil * bet);
    wkb[base + k] = to_bf16(wkv * bet);
    u[base + k]   = to_bf16(wkv * bet * de);
  }
}

// ---------------------------------------------------------------------------
// K5a: per (b,h,chunk): A_strict = -(wkb @ wk^T) .* L (strict lower)
//      and attn = (rk @ wk^T) .* L (inclusive lower).  64x64, K=128 WMMA.
// ---------------------------------------------------------------------------
__global__ __launch_bounds__(256) void chunk_attn_kernel(
    const bf16* __restrict__ rk, const bf16* __restrict__ wk,
    const bf16* __restrict__ wkb, const float* __restrict__ deccum,
    bf16* __restrict__ Abuf, bf16* __restrict__ attnbuf) {
  int bhc = blockIdx.x;
  __shared__ bf16 lhs[CC * KK];
  __shared__ bf16 wks[CC * KK];
  __shared__ float dec_s[CC];
  size_t base = (size_t)bhc * CC * KK;
  int tid = threadIdx.x, lane = tid & 31, wave = tid >> 5;
  for (int i = tid; i < CC * KK; i += 256) {
    lhs[i] = wkb[base + i];
    wks[i] = wk[base + i];
  }
  if (tid < CC) dec_s[tid] = deccum[(size_t)bhc * CC + tid];
  __syncthreads();
  for (int p = 0; p < 2; ++p) {
#pragma unroll
    for (int j = 0; j < 2; ++j) {
      int tile = wave * 2 + j;
      int mt = tile >> 2, nt = tile & 3;
      f32x8 acc = f32x8_zero();
#pragma unroll
      for (int ks = 0; ks < 4; ++ks) {
        bf16x16 a  = frag_a(&lhs[(mt * 16) * KK + ks * 32], KK, lane);
        bf16x16 bb = frag_bt(&wks[(nt * 16) * KK + ks * 32], KK, lane);
        acc = wmma_bf16(a, bb, acc);
      }
      int n = lane & 15, h2 = lane >> 4;
#pragma unroll
      for (int r = 0; r < 8; ++r) {
        int cc = mt * 16 + h2 * 8 + r, dd = nt * 16 + n;
        float lm = __expf(dec_s[cc] - dec_s[dd]);
        float val;
        if (p == 0) val = (cc > dd) ? -(acc[r] * lm) : 0.f;
        else        val = (cc >= dd) ? (acc[r] * lm) : 0.f;
        bf16* dst = (p == 0) ? Abuf : attnbuf;
        dst[(size_t)bhc * CC * CC + (size_t)cc * CC + dd] = to_bf16(val);
      }
    }
    __syncthreads();
    if (p == 0) {
      for (int i = tid; i < CC * KK; i += 256) lhs[i] = rk[base + i];
      __syncthreads();
    }
  }
}

// ---------------------------------------------------------------------------
// K5b: apply T = (I - A)^-1 to [v | u] via nilpotent doubling:
//      Y <- (I + A^(2^i)) Y for i=0..5 ; powers of A commute.
//      Emits vT = T v  and wkcneg = -(T u).
// ---------------------------------------------------------------------------
__global__ __launch_bounds__(256) void chunk_xform_kernel(
    const bf16* __restrict__ Abuf, const bf16* __restrict__ v,
    const bf16* __restrict__ u, bf16* __restrict__ vT,
    bf16* __restrict__ wkcneg) {
  int bhc = blockIdx.x;
  __shared__ bf16 Sa[CC * CC], Sb[CC * CC];
  __shared__ bf16 Y[CC * 256];
  int tid = threadIdx.x, lane = tid & 31, wave = tid >> 5;
  size_t baseK = (size_t)bhc * CC * KK, baseC = (size_t)bhc * CC * CC;
  for (int i = tid; i < CC * CC; i += 256) Sa[i] = Abuf[baseC + i];
  for (int i = tid; i < CC * KK; i += 256) {
    int c = i / KK, k = i % KK;
    Y[c * 256 + k]       = v[baseK + i];
    Y[c * 256 + 128 + k] = u[baseK + i];
  }
  __syncthreads();
  bf16* Scur = Sa;
  bf16* Snxt = Sb;
  for (int it = 0; it < 6; ++it) {
    f32x8 acc[8];
#pragma unroll
    for (int j = 0; j < 8; ++j) {
      int tile = wave * 8 + j;
      int mt = tile >> 4, nt = tile & 15;
      f32x8 cf = load_c(&Y[(mt * 16) * 256 + nt * 16], 256, lane);
#pragma unroll
      for (int ks = 0; ks < 2; ++ks) {
        bf16x16 a  = frag_a(&Scur[(mt * 16) * CC + ks * 32], CC, lane);
        bf16x16 bb = frag_b(&Y[(ks * 32) * 256 + nt * 16], 256, lane);
        cf = wmma_bf16(a, bb, cf);
      }
      acc[j] = cf;
    }
    __syncthreads();
#pragma unroll
    for (int j = 0; j < 8; ++j) {
      int tile = wave * 8 + j;
      int mt = tile >> 4, nt = tile & 15;
      store_c_bf(&Y[(mt * 16) * 256 + nt * 16], 256, lane, acc[j]);
    }
    __syncthreads();
    if (it < 5) {  // S <- S @ S
      f32x8 sac[2];
#pragma unroll
      for (int j = 0; j < 2; ++j) {
        int tile = wave * 2 + j;
        int mt = tile >> 2, nt = tile & 3;
        f32x8 cf = f32x8_zero();
#pragma unroll
        for (int ks = 0; ks < 2; ++ks) {
          bf16x16 a  = frag_a(&Scur[(mt * 16) * CC + ks * 32], CC, lane);
          bf16x16 bb = frag_b(&Scur[(ks * 32) * CC + nt * 16], CC, lane);
          cf = wmma_bf16(a, bb, cf);
        }
        sac[j] = cf;
      }
#pragma unroll
      for (int j = 0; j < 2; ++j) {
        int tile = wave * 2 + j;
        int mt = tile >> 2, nt = tile & 3;
        store_c_bf(&Snxt[(mt * 16) * CC + nt * 16], CC, lane, sac[j]);
      }
      __syncthreads();
      bf16* tsw = Scur; Scur = Snxt; Snxt = tsw;
    }
  }
  for (int i = tid; i < CC * KK; i += 256) {
    int c = i / KK, k = i % KK;
    vT[baseK + i]     = Y[c * 256 + k];
    wkcneg[baseK + i] = to_bf16(-(float)Y[c * 256 + 128 + k]);
  }
}

// ---------------------------------------------------------------------------
// K6: sequential chunk scan. One block per (b,h). fp32 state S (128x128)
//     lives in LDS across the 64 chunks; all matmuls are WMMA bf16 with
//     on-the-fly fp32->bf16 fragment conversion from LDS.
// ---------------------------------------------------------------------------
__global__ __launch_bounds__(256) void scan_kernel(
    const bf16* __restrict__ rk, const bf16* __restrict__ wk,
    const bf16* __restrict__ vT, const bf16* __restrict__ wkcneg,
    const bf16* __restrict__ attnb, const float* __restrict__ deccum,
    float* __restrict__ obuf) {
  int bh = blockIdx.x;
  int b = bh / NH, hh = bh % NH;
  __shared__ float Ss[KK * KK];     // 64 KB fp32 state
  __shared__ bf16 vnew[CC * KK];    // 16 KB
  __shared__ float dec_s[CC];
  __shared__ float dw_s[CC];
  int tid = threadIdx.x, lane = tid & 31, wave = tid >> 5;
  for (int i = tid; i < KK * KK; i += 256) Ss[i] = 0.f;
  __syncthreads();
  for (int ch = 0; ch < NCH; ++ch) {
    size_t cbase = (size_t)bh * NCH + ch;
    size_t baseK = cbase * CC * KK, baseC = cbase * CC * CC;
    if (tid < CC) dec_s[tid] = deccum[cbase * CC + tid];
    __syncthreads();
    if (tid < CC) dw_s[tid] = __expf(dec_s[CC - 1] - dec_s[tid]);
    __syncthreads();

    // step 1: v_new = vT + (-T u) @ S     (64x128, K=128)
    f32x8 acc[8];
#pragma unroll
    for (int j = 0; j < 4; ++j) {
      int tile = wave * 4 + j;
      int mt = tile >> 3, nt = tile & 7;
      f32x8 cf = load_c(&vT[baseK + (mt * 16) * KK + nt * 16], KK, lane);
#pragma unroll
      for (int ks = 0; ks < 4; ++ks) {
        bf16x16 a  = frag_a(&wkcneg[baseK + (mt * 16) * KK + ks * 32], KK, lane);
        bf16x16 bb = frag_b(&Ss[(ks * 32) * KK + nt * 16], KK, lane);
        cf = wmma_bf16(a, bb, cf);
      }
      acc[j] = cf;
    }
#pragma unroll
    for (int j = 0; j < 4; ++j) {
      int tile = wave * 4 + j;
      int mt = tile >> 3, nt = tile & 7;
      store_c_bf(&vnew[(mt * 16) * KK + nt * 16], KK, lane, acc[j]);
    }
    __syncthreads();

    // step 2: o = (rk * exp(dec)) @ S + attn @ v_new
#pragma unroll
    for (int j = 0; j < 4; ++j) {
      int tile = wave * 4 + j;
      int mt = tile >> 3, nt = tile & 7;
      f32x8 cf = f32x8_zero();
      float sc = __expf(dec_s[mt * 16 + (lane & 15)]);
#pragma unroll
      for (int ks = 0; ks < 4; ++ks) {
        bf16x16 a  = frag_a_scaled(&rk[baseK + (mt * 16) * KK + ks * 32], KK,
                                   lane, sc);
        bf16x16 bb = frag_b(&Ss[(ks * 32) * KK + nt * 16], KK, lane);
        cf = wmma_bf16(a, bb, cf);
      }
#pragma unroll
      for (int ks = 0; ks < 2; ++ks) {
        bf16x16 a  = frag_a(&attnb[baseC + (mt * 16) * CC + ks * 32], CC, lane);
        bf16x16 bb = frag_b(&vnew[(ks * 32) * KK + nt * 16], KK, lane);
        cf = wmma_bf16(a, bb, cf);
      }
      int n = lane & 15, h2 = lane >> 4;
#pragma unroll
      for (int r = 0; r < 8; ++r) {
        int c = mt * 16 + h2 * 8 + r, col = nt * 16 + n;
        int t = ch * CC + c;
        obuf[((size_t)b * TT + t) * DD + hh * KK + col] = cf[r];
      }
    }

    // step 3: S = S*exp(dec_last) + (wk .* dw)^T @ v_new   (128x128, K=64)
    float elast = __expf(dec_s[CC - 1]);
#pragma unroll
    for (int j = 0; j < 8; ++j) {
      int tile = wave * 8 + j;
      int mt = tile >> 3, nt = tile & 7;
      f32x8 cf = load_c_scaled(&Ss[(mt * 16) * KK + nt * 16], KK, lane, elast);
#pragma unroll
      for (int ks = 0; ks < 2; ++ks) {
        bf16x16 a  = frag_a_wkT(&wk[baseK], mt * 16, ks * 32, lane, dw_s);
        bf16x16 bb = frag_b(&vnew[(ks * 32) * KK + nt * 16], KK, lane);
        cf = wmma_bf16(a, bb, cf);
      }
      acc[j] = cf;
    }
    __syncthreads();  // all reads of Ss complete before overwrite
#pragma unroll
    for (int j = 0; j < 8; ++j) {
      int tile = wave * 8 + j;
      int mt = tile >> 3, nt = tile & 7;
      store_c_f32(&Ss[(mt * 16) * KK + nt * 16], KK, lane, acc[j]);
    }
    __syncthreads();
  }
}

// ---------------------------------------------------------------------------
// K7: per-head RMSNorm, gate with silu(x.Wg), emit bf16 for final GEMM.
// ---------------------------------------------------------------------------
__global__ __launch_bounds__(256) void normgate_kernel(
    const float* __restrict__ obuf, const float* __restrict__ xvg,
    const float* __restrict__ norm_w, bf16* __restrict__ og) {
  int tok = blockIdx.x;
  __shared__ float ss[NH];
  int tid = threadIdx.x;
  if (tid < NH) ss[tid] = 0.f;
  __syncthreads();
  int d0 = tid * 4;
  float o4[4], p = 0.f;
#pragma unroll
  for (int e = 0; e < 4; ++e) {
    o4[e] = obuf[(size_t)tok * DD + d0 + e];
    p += o4[e] * o4[e];
  }
  int hh = d0 >> 7;
  atomicAdd(&ss[hh], p);
  __syncthreads();
  float rs = rsqrtf(ss[hh] / (float)KK + 1e-5f);
#pragma unroll
  for (int e = 0; e < 4; ++e) {
    int d = d0 + e, k = d & 127;
    float g = xvg[(size_t)tok * 2048 + 1024 + d];
    float sg = g / (1.f + __expf(-g));
    og[(size_t)tok * DD + d] = to_bf16(o4[e] * rs * norm_w[k] * sg);
  }
}

// ---------------------------------------------------------------------------
// host-side launcher
// ---------------------------------------------------------------------------
extern "C" void kernel_launch(void* const* d_in, const int* in_sizes, int n_in,
                              void* d_out, int out_size, void* d_ws,
                              size_t ws_size, hipStream_t stream) {
  const float* x       = (const float*)d_in[0];
  const float* Wv      = (const float*)d_in[1];
  const float* Wg      = (const float*)d_in[2];
  const float* Wo      = (const float*)d_in[3];
  const float* Wb      = (const float*)d_in[4];
  const float* Wa      = (const float*)d_in[5];
  const float* dt_bias = (const float*)d_in[6];
  const float* A_log   = (const float*)d_in[7];
  const float* norm_w  = (const float*)d_in[8];
  const float* conv_w  = (const float*)d_in[9];
  const float* conv_b  = (const float*)d_in[10];
  float* out = (float*)d_out;

  char* ws = (char*)d_ws;
  size_t off = 0;
  auto alloc = [&](size_t bytes) -> void* {
    void* p = ws + off;
    off += (bytes + 255) & ~(size_t)255;
    return p;
  };
  const size_t TD = (size_t)BTOK * DD;
  bf16*  WvgT   = (bf16*)alloc((size_t)DD * 2048 * 2);
  bf16*  WoT    = (bf16*)alloc((size_t)DD * DD * 2);
  float* xvg    = (float*)alloc((size_t)BTOK * 2048 * 4);
  float* beta   = (float*)alloc((size_t)BTOK * NH * 4);
  float* deccum = (float*)alloc((size_t)BTOK * NH * 4);
  bf16*  rk     = (bf16*)alloc(TD * 2);
  bf16*  wk     = (bf16*)alloc(TD * 2);
  bf16*  v      = (bf16*)alloc(TD * 2);
  bf16*  wkb    = (bf16*)alloc(TD * 2);
  bf16*  u      = (bf16*)alloc(TD * 2);
  bf16*  Abuf   = (bf16*)alloc((size_t)NB * NH * NCH * CC * CC * 2);
  bf16*  attnb  = (bf16*)alloc((size_t)NB * NH * NCH * CC * CC * 2);
  bf16*  vT     = (bf16*)alloc(TD * 2);
  bf16*  wkcn   = (bf16*)alloc(TD * 2);
  float* obuf   = (float*)alloc(TD * 4);
  bf16*  og     = (bf16*)alloc(TD * 2);
  (void)ws_size; (void)n_in; (void)in_sizes; (void)out_size;

  // K1: pack weights (bf16, transposed, Wv|Wg fused)
  pack_weights_kernel<<<(DD * DD + 255) / 256, 256, 0, stream>>>(
      Wv, Wg, Wo, WvgT, WoT);
  // K2: xvg = x @ [Wv^T | Wg^T]   (M=16384, N=2048, K=1024)
  gemm_bf16_kernel<float><<<dim3(BTOK / 128, 2048 / 256), 256, 0, stream>>>(
      x, WvgT, xvg, BTOK, 2048, DD);
  // K3: beta / decay / per-chunk cumsum
  small_proj_kernel<<<NB * NCH, 256, 0, stream>>>(x, Wb, Wa, dt_bias, A_log,
                                                  beta, deccum);
  // K4: rk/wk/v/wkb/u in chunked bf16 layout
  qkv_kernel<<<BTOK, 256, 0, stream>>>(x, xvg, conv_w, conv_b, beta, deccum,
                                       rk, wk, v, wkb, u);
  // K5a: A_strict and attn per chunk
  chunk_attn_kernel<<<NB * NH * NCH, 256, 0, stream>>>(rk, wk, wkb, deccum,
                                                       Abuf, attnb);
  // K5b: apply (I-A)^-1 via nilpotent doubling
  chunk_xform_kernel<<<NB * NH * NCH, 256, 0, stream>>>(Abuf, v, u, vT, wkcn);
  // K6: sequential state scan (one WG per (b,h))
  scan_kernel<<<NB * NH, 256, 0, stream>>>(rk, wk, vT, wkcn, attnb, deccum,
                                           obuf);
  // K7: RMSNorm + gate -> bf16
  normgate_kernel<<<BTOK, 256, 0, stream>>>(obuf, xvg, norm_w, og);
  // K8: out = og @ Wo^T   (M=16384, N=1024, K=1024)
  gemm_bf16_kernel<bf16><<<dim3(BTOK / 128, DD / 256), 256, 0, stream>>>(
      og, WoT, out, BTOK, DD, DD);
}
